// RNNCell_Encoder_28621662060980
// MI455X (gfx1250) — matrix operand order repaired and checked by reading, not compiled, not run on hardware
//
#include <hip/hip_runtime.h>
#include <hip/hip_bf16.h>

typedef __attribute__((ext_vector_type(16))) __bf16 v16bf;
typedef __attribute__((ext_vector_type(8)))  __bf16 v8bf;
typedef __attribute__((ext_vector_type(8)))  float  v8f;

typedef int v4i_t __attribute__((vector_size(16)));
typedef __attribute__((address_space(1))) v4i_t GLv4i;  // global 128-bit
typedef __attribute__((address_space(3))) v4i_t LSv4i;  // LDS 128-bit

#define T_STEPS 512
#define BATCH   512
#define DIN     512
#define HID     512
#define KTOT    1024   // DIN + HID combined K

#define BM  128        // block tile rows (batch)
#define BN  64         // block tile cols (hidden)
#define KC  32         // K chunk per WMMA
#define NC  (KTOT / KC)
#define LDA 40         // padded LDS stride (bf16) -> 80B rows, conflict-free
#define LDB 40

// ---------------------------------------------------------------------------
// async global -> LDS (ASYNCcnt path); builtin if present, else inline asm
// ---------------------------------------------------------------------------
__device__ __forceinline__ void async_b128(const void* g, void* l) {
#if __has_builtin(__builtin_amdgcn_global_load_async_to_lds_b128)
  __builtin_amdgcn_global_load_async_to_lds_b128(
      (GLv4i*)(unsigned long long)(uintptr_t)g,
      (LSv4i*)(unsigned)(uintptr_t)l, 0, 0);
#else
  asm volatile("global_load_async_to_lds_b128 %0, %1, off"
               :: "v"((unsigned)(uintptr_t)l),
                  "v"((unsigned long long)(uintptr_t)g)
               : "memory");
#endif
}

__device__ __forceinline__ void wait_async0() {
#if __has_builtin(__builtin_amdgcn_s_wait_asynccnt)
  __builtin_amdgcn_s_wait_asynccnt(0);
#else
  asm volatile("s_wait_asynccnt 0x0" ::: "memory");
#endif
}

// ---------------------------------------------------------------------------
// prep: Wc[h][0..511] = bf16(W_ih[h][:]),  Wc[h][512..1023] = bf16(W_hh[h][:])
//       bias[h] = b_ih[h] + b_hh[h]
// ---------------------------------------------------------------------------
__global__ __launch_bounds__(256)
void prep_kernel(const float* __restrict__ W_ih, const float* __restrict__ W_hh,
                 const float* __restrict__ b_ih, const float* __restrict__ b_hh,
                 __bf16* __restrict__ Wc, float* __restrict__ bias) {
  const int h   = blockIdx.x;
  const int tid = threadIdx.x;
  for (int i = tid; i < DIN; i += 256) {
    Wc[(size_t)h * KTOT + i]       = (__bf16)W_ih[(size_t)h * DIN + i];
    Wc[(size_t)h * KTOT + DIN + i] = (__bf16)W_hh[(size_t)h * HID + i];
  }
  if (tid == 0) bias[h] = b_ih[h] + b_hh[h];
}

__global__ __launch_bounds__(256)
void zero_h(unsigned short* __restrict__ h) {
  h[(size_t)blockIdx.x * 256 + threadIdx.x] = 0;  // bf16(0)
}

// full f32 -> bf16 conversion of x_0 (one float4 per thread)
__global__ __launch_bounds__(256)
void conv_x(const float* __restrict__ x, __bf16* __restrict__ xb) {
  const int i = blockIdx.x * 256 + threadIdx.x;
  float4 f = ((const float4*)x)[i];
  union { __bf16 h[4]; uint2 u; } t;
  t.h[0] = (__bf16)f.x; t.h[1] = (__bf16)f.y;
  t.h[2] = (__bf16)f.z; t.h[3] = (__bf16)f.w;
  ((uint2*)xb)[i] = t.u;
}

// ---------------------------------------------------------------------------
// One recurrence step:
//   h_out[b,n] = tanh(bias[n] + sum_k [xb_t|h_prev][b,k] * Wc[n,k]),  K = 1024
// Block: 256 threads = 8 waves; block tile 128x64; wave tile 32x32.
// Double-buffered LDS fed by GLOBAL_LOAD_ASYNC_TO_LDS_B128.
// Also converts this block's share of x_{t+1} to bf16 for the next step.
// ---------------------------------------------------------------------------
__global__ __launch_bounds__(256)
void rnn_step(const __bf16* __restrict__ xb_t,    // [BATCH][DIN] bf16
              const __bf16* __restrict__ h_prev,  // [BATCH][HID] bf16
              const __bf16* __restrict__ Wc,      // [HID][KTOT]  bf16
              const float*  __restrict__ bias,    // [HID]
              __bf16*       __restrict__ h_out,   // [BATCH][HID] bf16
              float*        __restrict__ f_out,   // optional [BATCH][HID] f32
              const float*  __restrict__ x_next,  // optional [BATCH][DIN] f32
              __bf16*       __restrict__ xb_next) // optional [BATCH][DIN] bf16
{
  __shared__ __bf16 As[2][BM][LDA];
  __shared__ __bf16 Bs[2][BN][LDB];

  const int tid  = threadIdx.x;
  const int lane = tid & 31;
  const int wave = tid >> 5;       // 0..7
  const int wm   = wave & 3;       // 0..3 -> 32-row slab
  const int wn   = wave >> 2;      // 0..1 -> 32-col slab
  const int lr   = lane & 15;
  const int hi   = lane >> 4;      // half-wave select

  const int bm0 = blockIdx.y * BM; // batch rows
  const int bn0 = blockIdx.x * BN; // hidden cols

  // staging indices (3 async b128 per thread per chunk)
  const int arow = tid >> 1;           // 0..127
  const int akq  = (tid & 1) * 16;     // 0 or 16
  const int brow = tid >> 2;           // 0..63
  const int bkq  = (tid & 3) * 8;      // 0,8,16,24

  v8f acc[2][2] = {};

  auto issue = [&](int buf, int c) {
    const __bf16* Asrc = (c < NC / 2) ? xb_t : h_prev;
    const int kc = (c & (NC / 2 - 1)) * KC;
    const __bf16* ga = Asrc + (size_t)(bm0 + arow) * DIN + kc + akq;
    async_b128(ga,     &As[buf][arow][akq]);
    async_b128(ga + 8, &As[buf][arow][akq + 8]);
    const __bf16* gb = Wc + (size_t)(bn0 + brow) * KTOT + c * KC + bkq;
    async_b128(gb, &Bs[buf][brow][bkq]);
  };

  issue(0, 0);

  #pragma unroll 1
  for (int c = 0; c < NC; ++c) {
    const int cur = c & 1;
    wait_async0();        // chunk c resident in LDS
    __syncthreads();      // all waves' async writes visible; prev reads done
    if (c + 1 < NC) issue(cur ^ 1, c + 1);  // overlap next chunk with compute

    // ---- fragments per CDNA5 16-bit WMMA layouts ----
    // A (16x32): lanes 0-15 hold K {0..7,16..23}, lanes 16-31 hold K {8..15,24..31}
    v16bf afrag[2];
    #pragma unroll
    for (int mi = 0; mi < 2; ++mi) {
      const int r = wm * 32 + mi * 16 + lr;
      v8bf lo  = *(const v8bf*)&As[cur][r][hi * 8];
      v8bf hi8 = *(const v8bf*)&As[cur][r][16 + hi * 8];
      afrag[mi] = __builtin_shufflevector(lo, hi8, 0,1,2,3,4,5,6,7,8,9,10,11,12,13,14,15);
    }
    // B (32x16): lanes 0-15 hold K {0..7,16..23} for col n, lanes 16-31 hold K {8..15,24..31}
    v16bf bfrag[2];
    #pragma unroll
    for (int ni = 0; ni < 2; ++ni) {
      const int cix = wn * 32 + ni * 16 + lr;
      v8bf lo  = *(const v8bf*)&Bs[cur][cix][hi * 8];
      v8bf hi8 = *(const v8bf*)&Bs[cur][cix][16 + hi * 8];
      bfrag[ni] = __builtin_shufflevector(lo, hi8, 0,1,2,3,4,5,6,7,8,9,10,11,12,13,14,15);
    }

    #pragma unroll
    for (int mi = 0; mi < 2; ++mi)
      #pragma unroll
      for (int ni = 0; ni < 2; ++ni)
        acc[mi][ni] = __builtin_amdgcn_wmma_f32_16x16x32_bf16(
            false, afrag[mi], false, bfrag[ni],
            (short)0, acc[mi][ni], false, false);
  }

  // ---- pipelined conversion of x_{t+1} (this block's 1/32 share) ----
  if (xb_next) {
    const int blk = blockIdx.y * gridDim.x + blockIdx.x;          // 0..31
    const size_t base = (size_t)blk * ((size_t)BATCH * DIN / 32); // 8192 elems
    const float4* s = (const float4*)(x_next + base);
    uint2* d = (uint2*)(xb_next + base);
    #pragma unroll 1
    for (int i = tid; i < (BATCH * DIN / 32) / 4; i += 256) {
      float4 f = s[i];
      union { __bf16 h[4]; uint2 u; } t;
      t.h[0] = (__bf16)f.x; t.h[1] = (__bf16)f.y;
      t.h[2] = (__bf16)f.z; t.h[3] = (__bf16)f.w;
      d[i] = t.u;
    }
  }

  // ---- epilogue: bias + tanh, store bf16 h (and f32 on final step) ----
  // C/D layout: lane (hi,lr): n = lr, m = hi*8 + vgpr_index
  #pragma unroll
  for (int mi = 0; mi < 2; ++mi) {
    #pragma unroll
    for (int ni = 0; ni < 2; ++ni) {
      const int n  = bn0 + wn * 32 + ni * 16 + lr;
      const float bv = bias[n];
      #pragma unroll
      for (int r = 0; r < 8; ++r) {
        const int m = bm0 + wm * 32 + mi * 16 + hi * 8 + r;
        const float v = tanhf(acc[mi][ni][r] + bv);
        h_out[(size_t)m * HID + n] = (__bf16)v;
        if (f_out) f_out[(size_t)m * HID + n] = v;
      }
    }
  }
}

// ---------------------------------------------------------------------------
extern "C" void kernel_launch(void* const* d_in, const int* in_sizes, int n_in,
                              void* d_out, int out_size, void* d_ws, size_t ws_size,
                              hipStream_t stream) {
  const float* x   = (const float*)d_in[0];  // [T,B,D]
  const float* Wih = (const float*)d_in[1];  // [H,D]
  const float* Whh = (const float*)d_in[2];  // [H,H]
  const float* bih = (const float*)d_in[3];  // [H]
  const float* bhh = (const float*)d_in[4];  // [H]
  (void)in_sizes; (void)n_in; (void)out_size; (void)ws_size;

  char* ws = (char*)d_ws;
  size_t off = 0;
  __bf16* Wc   = (__bf16*)(ws + off); off += (size_t)HID * KTOT * 2;  // 1 MB
  float*  bias = (float*)(ws + off);  off += 4096;                    // 2 KB used
  __bf16* hA   = (__bf16*)(ws + off); off += (size_t)BATCH * HID * 2; // 512 KB
  __bf16* hB   = (__bf16*)(ws + off); off += (size_t)BATCH * HID * 2; // 512 KB
  __bf16* xbA  = (__bf16*)(ws + off); off += (size_t)BATCH * DIN * 2; // 512 KB
  __bf16* xbB  = (__bf16*)(ws + off); off += (size_t)BATCH * DIN * 2; // 512 KB

  prep_kernel<<<HID, 256, 0, stream>>>(Wih, Whh, bih, bhh, Wc, bias);
  zero_h<<<(BATCH * HID) / 256, 256, 0, stream>>>((unsigned short*)hA);
  conv_x<<<(BATCH * DIN / 4) / 256, 256, 0, stream>>>(x, xbA);  // x_0

  dim3 grid(HID / BN, BATCH / BM);   // (8, 4) = 32 blocks per step
  __bf16* hin  = hA;
  __bf16* hout = hB;
  for (int t = 0; t < T_STEPS; ++t) {
    const bool last = (t == T_STEPS - 1);
    __bf16* xb_cur = (t & 1) ? xbB : xbA;
    __bf16* xb_nxt = (t & 1) ? xbA : xbB;
    rnn_step<<<grid, 256, 0, stream>>>(
        xb_cur, hin, Wc, bias, hout,
        last ? (float*)d_out : nullptr,
        last ? nullptr : x + (size_t)(t + 1) * BATCH * DIN,
        last ? nullptr : xb_nxt);
    __bf16* tmp = hin; hin = hout; hout = tmp;
  }
}
